// ResponseFilter_60000693125186
// MI455X (gfx1250) — matrix operand order repaired and checked by reading, not compiled, not run on hardware
//
#include <hip/hip_runtime.h>
#include <hip/hip_bf16.h>

// ---------------- problem constants ----------------
#define Bsz   16
#define Tsz   32
#define Lsz   64
#define Dsz   768
#define Hsz   12
#define HDsz  64
#define Fsz   2048
#define Zsz   100
#define ZPsz  128          // Z padded to multiple of 32 for WMMA K
#define NLn   2
#define NMAX  (Bsz*(Tsz-1))   // 496 packed replies max
#define MMAX  (NMAX*Lsz)      // 31744 rows max
#define NPAD  512             // padded key count for attention
#define EPSv  1e-5f
#define FINF  __builtin_inff()

typedef __attribute__((ext_vector_type(16))) __bf16       v16bf;
typedef __attribute__((ext_vector_type(8)))  float        v8f;
typedef __attribute__((ext_vector_type(4)))  unsigned int u32x4;
typedef __attribute__((ext_vector_type(4)))  int          v4i;

union BF16x16 {
  v16bf          bf;
  u32x4          q[2];
  unsigned short s[16];
};

// ---------------- async global->LDS staging (CDNA5 GLOBAL_LOAD_ASYNC_TO_LDS_B128) ----------
#if __has_builtin(__builtin_amdgcn_global_load_async_to_lds_b128) && \
    __has_builtin(__builtin_amdgcn_s_wait_asynccnt)
#define USE_ASYNC_LDS 1
#else
#define USE_ASYNC_LDS 0
#endif

// 16-byte global -> LDS copy. Async path writes LDS directly (ASYNCcnt-tracked),
// bypassing VGPRs; fallback is the synchronous load/store pair.
// Builtin signature (from compiler diagnostic): (v4i AS1*, v4i AS3*, imm offset, imm cpol).
__device__ __forceinline__ void lds_cp16(void* lds, const void* g) {
#if USE_ASYNC_LDS
  __builtin_amdgcn_global_load_async_to_lds_b128(
      (__attribute__((address_space(1))) v4i*)(size_t)g,
      (__attribute__((address_space(3))) v4i*)(size_t)lds,  // flat LDS addr truncates to LDS offset (ISA 10.2)
      0, 0);
#else
  *(u32x4*)lds = *(const u32x4*)g;
#endif
}
__device__ __forceinline__ void lds_zero16(void* lds) {
  *(u32x4*)lds = (u32x4)0u;
}
// Each wave must drain its own ASYNCcnt before the workgroup barrier.
__device__ __forceinline__ void async_fence() {
#if USE_ASYNC_LDS
  __builtin_amdgcn_s_wait_asynccnt(0);
#endif
}

// meta layout (int offsets inside workspace)
#define MI_NTOT  0
#define MI_NRESP 1
#define MI_NEXT  17
#define MI_RESPB 64
#define MI_RESPT (64 + NMAX)
#define MI_ORDER (64 + 2*NMAX)               // B*(T-1)
#define MI_NODE  (MI_ORDER + Bsz*(Tsz-1))    // B*T

__device__ __forceinline__ unsigned short f2bf(float f) {
  unsigned u = __float_as_uint(f);
  unsigned r = u + 0x7FFFu + ((u >> 16) & 1u);  // round-to-nearest-even
  return (unsigned short)(r >> 16);
}

// ---------------- setup: ragged packing metadata ----------------
__global__ void k_setup(const int* __restrict__ tree_lens, int* __restrict__ meta) {
  if (threadIdx.x == 0 && blockIdx.x == 0) {
    int tot = 0;
    for (int b = 0; b < Bsz; ++b) {
      int tl = tree_lens[b];
      int nr = tl - 1;
      meta[MI_NRESP + b] = nr;
      int ne = nr / 20;              // floor(nr*0.05)
      if (ne < 1) ne = 1;
      meta[MI_NEXT + b] = ne;
      for (int t = 1; t < tl; ++t) { meta[MI_RESPB + tot] = b; meta[MI_RESPT + tot] = t; ++tot; }
    }
    meta[MI_NTOT] = tot;
  }
}

// ---------------- pack replies -> xf (f32), xb (bf16), maskp ----------------
__global__ __launch_bounds__(256)
void k_pack(const float* __restrict__ emb, const float* __restrict__ amask,
            const int* __restrict__ meta, float* __restrict__ xf,
            unsigned short* __restrict__ xb, float* __restrict__ maskp) {
  const int n = blockIdx.x, l = blockIdx.y;
  if (n >= meta[MI_NTOT]) return;
  const int b = meta[MI_RESPB + n], t = meta[MI_RESPT + n];
  const float* src = emb + (((size_t)b*Tsz + t)*Lsz + l)*Dsz;
  float*          df = xf + ((size_t)n*Lsz + l)*Dsz;
  unsigned short* db = xb + ((size_t)n*Lsz + l)*Dsz;
  for (int i = threadIdx.x; i < Dsz; i += 256) { float v = src[i]; df[i] = v; db[i] = f2bf(v); }
  if (threadIdx.x == 0) maskp[n*Lsz + l] = amask[((size_t)b*Tsz + t)*Lsz + l];
}

// ---------------- weight f32 -> bf16 cast with optional col padding ----------------
__global__ __launch_bounds__(256)
void k_castpad(const float* __restrict__ src, unsigned short* __restrict__ dst,
               int rows, int cols, int ldd) {
  size_t i = (size_t)blockIdx.x*256 + threadIdx.x;
  size_t tot = (size_t)rows * ldd;
  if (i >= tot) return;
  int r = (int)(i / ldd), c = (int)(i % ldd);
  dst[i] = (c < cols) ? f2bf(src[(size_t)r*cols + c]) : (unsigned short)0;
}

// ---------------- WMMA bf16 GEMM: C[M,NP] = act(A[M,K] * W[N,K]^T + bias) ----------------
// block 256 = 8 waves; 128x128 C tile; wave -> 16x128 strip; K step 32 via LDS,
// staged with async global->LDS copies when available.
__global__ __launch_bounds__(256)
void k_gemm(const unsigned short* __restrict__ A, const unsigned short* __restrict__ Wm,
            const float* __restrict__ bias, const int* __restrict__ meta,
            int K, int N, int NP, float* __restrict__ Cf, unsigned short* __restrict__ Cb,
            int act /*0 none,1 relu,2 tanh*/) {
  const int M  = meta[MI_NTOT] * Lsz;
  const int m0 = blockIdx.y * 128;
  const int n0 = blockIdx.x * 128;
  if (m0 >= M) return;

  __shared__ unsigned short As[128*32];
  __shared__ unsigned short Ws[128*32];

  const int tid  = threadIdx.x;
  const int wave = tid >> 5, lane = tid & 31;
  const int lm = lane & 15, lg = lane >> 4;

  v8f acc[8];
  for (int j = 0; j < 8; ++j) acc[j] = (v8f)0.0f;

  const int row = tid >> 1, half = tid & 1;
  const int kTiles = (K + 31) >> 5;

  for (int kt = 0; kt < kTiles; ++kt) {
    const int k0 = kt << 5;
    const int gk = k0 + half*16;
    {   // stage A tile (128x32)
      const int gm = m0 + row;
      unsigned short* dst = &As[row*32 + half*16];
      if (gm < M) {
        if ((K & 31) == 0) {
          const unsigned short* src = A + (size_t)gm*K + gk;
          lds_cp16(dst,     src);
          lds_cp16(dst + 8, src + 8);
          if (kt + 1 < kTiles) __builtin_prefetch(A + (size_t)gm*K + gk + 32, 0, 1);
        } else {
          for (int i2 = 0; i2 < 16; ++i2) { int k = gk + i2; dst[i2] = (k < K) ? A[(size_t)gm*K + k] : (unsigned short)0; }
        }
      } else { lds_zero16(dst); lds_zero16(dst + 8); }
    }
    {   // stage W tile (128x32)
      const int gn = n0 + row;
      unsigned short* dst = &Ws[row*32 + half*16];
      if (gn < N) {
        if ((K & 31) == 0) {
          const unsigned short* src = Wm + (size_t)gn*K + gk;
          lds_cp16(dst,     src);
          lds_cp16(dst + 8, src + 8);
        } else {
          for (int i2 = 0; i2 < 16; ++i2) { int k = gk + i2; dst[i2] = (k < K) ? Wm[(size_t)gn*K + k] : (unsigned short)0; }
        }
      } else { lds_zero16(dst); lds_zero16(dst + 8); }
    }
    async_fence();
    __syncthreads();

    // A fragment: lane holds row lm, 16 contiguous K at lg*16
    BF16x16 afr;
    const unsigned short* ap = &As[(wave*16 + lm)*32 + lg*16];
    afr.q[0] = *(const u32x4*)ap;
    afr.q[1] = *(const u32x4*)(ap + 8);
    for (int j = 0; j < 8; ++j) {
      BF16x16 bfr;   // B fragment: lane holds column lm, 16 contiguous K at lg*16
      const unsigned short* bp = &Ws[(j*16 + lm)*32 + lg*16];
      bfr.q[0] = *(const u32x4*)bp;
      bfr.q[1] = *(const u32x4*)(bp + 8);
      acc[j] = __builtin_amdgcn_wmma_f32_16x16x32_bf16(false, afr.bf, false, bfr.bf,
                                                       (short)0, acc[j], false, false);
    }
    __syncthreads();
  }

  for (int j = 0; j < 8; ++j) {
    const int col = n0 + j*16 + lm;
    if (col >= NP) continue;
    const float bv = (bias && col < N) ? bias[col] : 0.0f;
    for (int v = 0; v < 8; ++v) {
      const int grow = m0 + wave*16 + v + 8*lg;   // C/D layout: row v+8*(lane/16), col lane%16
      if (grow >= M) continue;
      float val;
      if (col < N) {
        val = acc[j][v] + bv;
        if (act == 1)      val = fmaxf(val, 0.0f);
        else if (act == 2) val = tanhf(val);
      } else val = 0.0f;   // zero the N..NP pad (keeps padded-K GEMMs exact)
      if (Cf) Cf[(size_t)grow*NP + col] = val;
      if (Cb) Cb[(size_t)grow*NP + col] = f2bf(val);
    }
  }
}

// ---------------- fused attention: per (token l, head h, 16-query strip) ----------------
__global__ __launch_bounds__(32)
void k_attn(const unsigned short* __restrict__ qkv, const float* __restrict__ maskp,
            const int* __restrict__ meta, unsigned short* __restrict__ ob) {
  const int lq = blockIdx.x / Hsz;
  const int h  = blockIdx.x % Hsz;
  const int Ntot  = meta[MI_NTOT];
  const int nbase = blockIdx.y * 16;
  if (nbase >= Ntot) return;

  __shared__ unsigned short Qs[16*64];
  __shared__ unsigned short Ks[16*64];
  __shared__ unsigned short Vs[32*64];
  __shared__ float          Sc[16*NPAD];
  __shared__ unsigned short Ps[16*NPAD];

  const int lane = threadIdx.x;
  const int lm = lane & 15, lg = lane >> 4;
  const int row3D = 3*Dsz;

  { // load Q strip 16x64 (async staged)
    const int m = lane >> 1, half = lane & 1;
    const int n = nbase + m;
    unsigned short* dst = &Qs[m*64 + half*32];
    if (n < Ntot) {
      const unsigned short* src = qkv + ((size_t)n*Lsz + lq)*row3D + h*HDsz + half*32;
      lds_cp16(dst, src); lds_cp16(dst + 8, src + 8);
    } else { lds_zero16(dst); lds_zero16(dst + 8); }
  }
  async_fence();
  __syncthreads();

  // scores: S[16, Ntot] = (Q K^T)/8 + float-mask bias
  const int nkt = (Ntot + 15) >> 4;
  for (int kt = 0; kt < nkt; ++kt) {
    {
      const int m = lane >> 1, half = lane & 1;
      const int n = kt*16 + m;
      unsigned short* dst = &Ks[m*64 + half*32];
      if (n < Ntot) {
        const unsigned short* src = qkv + ((size_t)n*Lsz + lq)*row3D + Dsz + h*HDsz + half*32;
        lds_cp16(dst, src); lds_cp16(dst + 8, src + 8);
      } else { lds_zero16(dst); lds_zero16(dst + 8); }
    }
    async_fence();
    __syncthreads();
    v8f c = (v8f)0.0f;
    for (int kc = 0; kc < 2; ++kc) {       // hd=64 -> two K=32 WMMAs
      BF16x16 afr, bfr;
      const unsigned short* ap = &Qs[lm*64 + kc*32 + lg*16];
      afr.q[0] = *(const u32x4*)ap; afr.q[1] = *(const u32x4*)(ap + 8);
      const unsigned short* bp = &Ks[lm*64 + kc*32 + lg*16];
      bfr.q[0] = *(const u32x4*)bp; bfr.q[1] = *(const u32x4*)(bp + 8);
      c = __builtin_amdgcn_wmma_f32_16x16x32_bf16(false, afr.bf, false, bfr.bf,
                                                  (short)0, c, false, false);
    }
    const int key = kt*16 + lm;
    const float bias = (key < Ntot) ? maskp[key*Lsz + lq] : 0.0f;
    for (int v = 0; v < 8; ++v) {
      const int r = v + 8*lg;
      Sc[r*NPAD + key] = (key < Ntot) ? (c[v]*0.125f + bias) : -FINF;
    }
    __syncthreads();
  }
  for (int r = 0; r < 16; ++r)
    for (int c2 = nkt*16 + lane; c2 < NPAD; c2 += 32) Sc[r*NPAD + c2] = -FINF;
  __syncthreads();

  // wave softmax per query row -> Ps (bf16)
  for (int r = 0; r < 16; ++r) {
    float mx = -FINF;
    for (int c2 = lane; c2 < NPAD; c2 += 32) mx = fmaxf(mx, Sc[r*NPAD + c2]);
    for (int off = 16; off; off >>= 1) mx = fmaxf(mx, __shfl_xor(mx, off, 32));
    float sum = 0.0f;
    for (int c2 = lane; c2 < NPAD; c2 += 32) {
      float e = __expf(Sc[r*NPAD + c2] - mx);
      Sc[r*NPAD + c2] = e;
      sum += e;
    }
    for (int off = 16; off; off >>= 1) sum += __shfl_xor(sum, off, 32);
    const float inv = (sum > 0.0f) ? 1.0f / sum : 0.0f;
    for (int c2 = lane; c2 < NPAD; c2 += 32) Ps[r*NPAD + c2] = f2bf(Sc[r*NPAD + c2] * inv);
  }
  __syncthreads();

  // O = P V, 32-key chunks
  v8f oacc[4];
  for (int j = 0; j < 4; ++j) oacc[j] = (v8f)0.0f;
  const int nkc = (Ntot + 31) >> 5;
  for (int kc = 0; kc < nkc; ++kc) {
    {
      const int key = kc*32 + lane;
      unsigned short* dst = &Vs[lane*64];
      if (key < Ntot) {
        const unsigned short* src = qkv + ((size_t)key*Lsz + lq)*row3D + 2*Dsz + h*HDsz;
        lds_cp16(dst,      src);
        lds_cp16(dst + 8,  src + 8);
        lds_cp16(dst + 16, src + 16);
        lds_cp16(dst + 24, src + 24);
      } else {
        lds_zero16(dst); lds_zero16(dst + 8); lds_zero16(dst + 16); lds_zero16(dst + 24);
      }
    }
    async_fence();
    __syncthreads();
    BF16x16 afr;
    const unsigned short* ap = &Ps[lm*NPAD + kc*32 + lg*16];
    afr.q[0] = *(const u32x4*)ap; afr.q[1] = *(const u32x4*)(ap + 8);
    for (int j = 0; j < 4; ++j) {
      BF16x16 bfr;
      const int ch = j*16 + lm;
      for (int i = 0; i < 16; ++i) bfr.s[i] = Vs[(lg*16 + i)*64 + ch];
      oacc[j] = __builtin_amdgcn_wmma_f32_16x16x32_bf16(false, afr.bf, false, bfr.bf,
                                                        (short)0, oacc[j], false, false);
    }
    __syncthreads();
  }
  for (int j = 0; j < 4; ++j) {
    const int ch = h*HDsz + j*16 + lm;
    for (int v = 0; v < 8; ++v) {
      const int n = nbase + v + 8*lg;
      if (n < Ntot) ob[((size_t)n*Lsz + lq)*Dsz + ch] = f2bf(oacc[j][v]);
    }
  }
}

// ---------------- residual + LayerNorm (in-place ok) ----------------
__global__ __launch_bounds__(256)
void k_lnres(const float* __restrict__ x, const float* __restrict__ a,
             const float* __restrict__ g, const float* __restrict__ bta,
             const int* __restrict__ meta, float* __restrict__ yf,
             unsigned short* __restrict__ yb) {
  const int M = meta[MI_NTOT] * Lsz;
  const int row = blockIdx.x;
  if (row >= M) return;
  __shared__ float red[256];
  const float* xp = x + (size_t)row*Dsz;
  const float* ap = a + (size_t)row*Dsz;
  float vals[3], lsum = 0.0f;
  for (int i = 0; i < 3; ++i) { int d = threadIdx.x + i*256; float v = xp[d] + ap[d]; vals[i] = v; lsum += v; }
  red[threadIdx.x] = lsum; __syncthreads();
  for (int s = 128; s; s >>= 1) { if (threadIdx.x < s) red[threadIdx.x] += red[threadIdx.x + s]; __syncthreads(); }
  const float mean = red[0] / Dsz; __syncthreads();
  float lv = 0.0f;
  for (int i = 0; i < 3; ++i) { float dd = vals[i] - mean; lv += dd*dd; }
  red[threadIdx.x] = lv; __syncthreads();
  for (int s = 128; s; s >>= 1) { if (threadIdx.x < s) red[threadIdx.x] += red[threadIdx.x + s]; __syncthreads(); }
  const float rstd = rsqrtf(red[0] / Dsz + EPSv);
  for (int i = 0; i < 3; ++i) {
    int d = threadIdx.x + i*256;
    float o = (vals[i] - mean)*rstd*g[d] + bta[d];
    yf[(size_t)row*Dsz + d] = o;
    yb[(size_t)row*Dsz + d] = f2bf(o);
  }
}

// ---------------- per-response mean of original embeddings ----------------
__global__ __launch_bounds__(256)
void k_embmean(const float* __restrict__ xf, const int* __restrict__ meta,
               float* __restrict__ emean) {
  const int n = blockIdx.x;
  if (n >= meta[MI_NTOT]) return;
  for (int d = threadIdx.x; d < Dsz; d += 256) {
    float s = 0.0f;
    for (int l = 0; l < Lsz; ++l) s += xf[((size_t)n*Lsz + l)*Dsz + d];
    emean[(size_t)n*Dsz + d] = s * (1.0f/Lsz);
  }
}

// ---------------- anomaly scores: sum_D (mean_L d - mean_L emb)^2 ----------------
__global__ __launch_bounds__(256)
void k_score(const float* __restrict__ df, const float* __restrict__ emean,
             const int* __restrict__ meta, float* __restrict__ scores) {
  const int n = blockIdx.x;
  __shared__ float red[256];
  if (n >= meta[MI_NTOT]) { if (threadIdx.x == 0) scores[n] = 0.0f; return; }
  float acc = 0.0f;
  for (int d = threadIdx.x; d < Dsz; d += 256) {
    float s = 0.0f;
    for (int l = 0; l < Lsz; ++l) s += df[((size_t)n*Lsz + l)*Dsz + d];
    float diff = s*(1.0f/Lsz) - emean[(size_t)n*Dsz + d];
    acc += diff*diff;
  }
  red[threadIdx.x] = acc; __syncthreads();
  for (int s = 128; s; s >>= 1) { if (threadIdx.x < s) red[threadIdx.x] += red[threadIdx.x + s]; __syncthreads(); }
  if (threadIdx.x == 0) scores[n] = red[0];
}

// ---------------- per-batch stable top-(T-1) selection (reference's prefix-argsort quirk) ----
__global__ __launch_bounds__(256)
void k_select(const float* __restrict__ scores, int* __restrict__ meta) {
  const int b = blockIdx.x;
  const int nr = meta[MI_NRESP + b];
  __shared__ float sv[NMAX];
  __shared__ float rv[256];
  __shared__ int   ri[256];
  for (int i = threadIdx.x; i < NMAX; i += 256) sv[i] = (i < nr) ? scores[i] : FINF;
  __syncthreads();
  for (int j = 0; j < Tsz - 1; ++j) {
    float best = FINF; int bi = NMAX;
    for (int i = threadIdx.x; i < NMAX; i += 256) {
      float v = sv[i];
      if (v < best || (v == best && i < bi)) { best = v; bi = i; }
    }
    rv[threadIdx.x] = best; ri[threadIdx.x] = bi; __syncthreads();
    for (int s = 128; s; s >>= 1) {
      if (threadIdx.x < s) {
        if (rv[threadIdx.x + s] < rv[threadIdx.x] ||
            (rv[threadIdx.x + s] == rv[threadIdx.x] && ri[threadIdx.x + s] < ri[threadIdx.x])) {
          rv[threadIdx.x] = rv[threadIdx.x + s]; ri[threadIdx.x] = ri[threadIdx.x + s];
        }
      }
      __syncthreads();
    }
    if (threadIdx.x == 0) {
      int win = ri[0];
      meta[MI_ORDER + b*(Tsz-1) + j] = win;
      if (win < NMAX) sv[win] = __builtin_nanf("");   // exclude from further rounds
    }
    __syncthreads();
  }
  if (threadIdx.x == 0) {
    const int ne = meta[MI_NEXT + b];
    meta[MI_NODE + b*Tsz + 0] = 0;
    for (int t = 1; t < Tsz; ++t) {
      int j = t - 1;
      meta[MI_NODE + b*Tsz + t] = (j < ne) ? meta[MI_ORDER + b*(Tsz-1) + j] + 1 : 0;
    }
  }
}

// ---------------- small outputs: oh, ext_mask, n_ext_adv, new_tree_lens ----------------
__global__ __launch_bounds__(256)
void k_outsmall(const int* __restrict__ meta, float* __restrict__ oh,
                float* __restrict__ extm, float* __restrict__ nadv,
                float* __restrict__ ntl) {
  const int b = blockIdx.x;
  const int ne = meta[MI_NEXT + b];
  __shared__ int ohl[Tsz];
  for (int t = threadIdx.x; t < Tsz; t += 256) ohl[t] = 0;
  __syncthreads();
  if (threadIdx.x == 0) {
    ohl[0] = 1;
    for (int j = 0; j < ne && j < Tsz - 1; ++j) {
      int idx = meta[MI_ORDER + b*(Tsz-1) + j] + 1;
      if (idx < Tsz) ohl[idx] = 1;
    }
  }
  __syncthreads();
  for (int t = threadIdx.x; t < Tsz; t += 256) oh[b*Tsz + t] = (float)ohl[t];
  for (int i = threadIdx.x; i < Tsz*Lsz; i += 256) extm[b*Tsz*Lsz + i] = (float)ohl[i / Lsz];
  if (b == 0 && threadIdx.x == 0) nadv[0] = 0.0f;
  if (threadIdx.x == 0) ntl[b] = (float)(1 + ne);
}

// ---------------- gather new_emb / new_msk ----------------
__global__ __launch_bounds__(256)
void k_gather(const int* __restrict__ meta, const float* __restrict__ emb,
              const float* __restrict__ amask, float* __restrict__ new_emb,
              float* __restrict__ new_msk) {
  const int bt = blockIdx.x, l = blockIdx.y;
  const int b = bt / Tsz, t = bt % Tsz;
  const int node = meta[MI_NODE + b*Tsz + t];
  const int ne = meta[MI_NEXT + b];
  const bool valid = (t == 0) || (t - 1 < ne);
  const float* src = emb + (((size_t)b*Tsz + node)*Lsz + l)*Dsz;
  float* dst = new_emb + (((size_t)bt)*Lsz + l)*Dsz;
  for (int i = threadIdx.x; i < Dsz; i += 256) dst[i] = src[i];
  if (threadIdx.x == 0)
    new_msk[(size_t)bt*Lsz + l] = amask[((size_t)b*Tsz + node)*Lsz + l] * (valid ? 1.0f : 0.0f);
}

// ================= host orchestration =================
extern "C" void kernel_launch(void* const* d_in, const int* in_sizes, int n_in,
                              void* d_out, int out_size, void* d_ws, size_t ws_size,
                              hipStream_t stream) {
  (void)in_sizes; (void)n_in; (void)out_size; (void)ws_size;

  const int*   tlens = (const int*)d_in[0];
  const float* emb   = (const float*)d_in[1];
  const float* amask = (const float*)d_in[2];
  const float* pe_w  = (const float*)d_in[27];
  const float* pe_b  = (const float*)d_in[28];
  const float* pd_w  = (const float*)d_in[29];
  const float* pd_b  = (const float*)d_in[30];

  size_t cur = 0;
  auto alloc = [&](size_t bytes) -> char* {
    char* p = (char*)d_ws + cur;
    cur += (bytes + 255) & ~(size_t)255;
    return p;
  };

  int*            meta  = (int*)alloc(4096 * sizeof(int));
  float*          xf    = (float*)alloc((size_t)MMAX*Dsz*4);
  float*          af    = (float*)alloc((size_t)MMAX*Dsz*4);
  unsigned short* xb    = (unsigned short*)alloc((size_t)MMAX*Dsz*2);
  unsigned short* qkvb  = (unsigned short*)alloc((size_t)MMAX*3*Dsz*2);
  unsigned short* ob    = (unsigned short*)alloc((size_t)MMAX*Dsz*2);
  unsigned short* fb    = (unsigned short*)alloc((size_t)MMAX*Fsz*2);
  unsigned short* zb    = (unsigned short*)alloc((size_t)MMAX*ZPsz*2);
  float*          maskp = (float*)alloc((size_t)NMAX*Lsz*4);
  float*          emean = (float*)alloc((size_t)NMAX*Dsz*4);
  float*          scores= (float*)alloc((size_t)NMAX*4);

  auto castpad = [&](const float* src, unsigned short* dst, int rows, int cols, int ldd) {
    size_t tot = (size_t)rows * ldd;
    k_castpad<<<dim3((unsigned)((tot + 255)/256)), 256, 0, stream>>>(src, dst, rows, cols, ldd);
  };

  // bf16 weight mirrors per stack (enc=0, dec=1)
  unsigned short* wQ[2]; unsigned short* wO[2]; unsigned short* w1b[2]; unsigned short* w2b[2];
  const float *bQ[2], *bO[2], *g1[2], *b1[2], *c1[2], *c2[2], *g2[2], *b2[2];
  for (int s = 0; s < 2; ++s) {
    const int base = 3 + s*12;
    const float* qkvw = (const float*)d_in[base + 0];
    bQ[s] = (const float*)d_in[base + 1];
    const float* outw = (const float*)d_in[base + 2];
    bO[s] = (const float*)d_in[base + 3];
    g1[s] = (const float*)d_in[base + 4];
    b1[s] = (const float*)d_in[base + 5];
    const float* w1 = (const float*)d_in[base + 6];
    c1[s] = (const float*)d_in[base + 7];
    const float* w2 = (const float*)d_in[base + 8];
    c2[s] = (const float*)d_in[base + 9];
    g2[s] = (const float*)d_in[base + 10];
    b2[s] = (const float*)d_in[base + 11];

    wQ[s]  = (unsigned short*)alloc((size_t)NLn*3*Dsz*Dsz*2);
    wO[s]  = (unsigned short*)alloc((size_t)NLn*Dsz*Dsz*2);
    w1b[s] = (unsigned short*)alloc((size_t)NLn*Fsz*Dsz*2);
    w2b[s] = (unsigned short*)alloc((size_t)NLn*Dsz*Fsz*2);
    castpad(qkvw, wQ[s],  NLn*3*Dsz, Dsz, Dsz);
    castpad(outw, wO[s],  NLn*Dsz,   Dsz, Dsz);
    castpad(w1,   w1b[s], NLn*Fsz,   Dsz, Dsz);
    castpad(w2,   w2b[s], NLn*Dsz,   Fsz, Fsz);
  }
  unsigned short* pewb = (unsigned short*)alloc((size_t)Zsz*Dsz*2);
  unsigned short* pdwb = (unsigned short*)alloc((size_t)Dsz*ZPsz*2);
  castpad(pe_w, pewb, Zsz, Dsz, Dsz);
  castpad(pd_w, pdwb, Dsz, Zsz, ZPsz);   // pad K 100 -> 128 with zeros

  auto gemm = [&](const unsigned short* A, const unsigned short* Wm, const float* bias,
                  int K, int N, int NP, float* Cf, unsigned short* Cb, int act) {
    dim3 g((unsigned)((NP + 127)/128), (unsigned)(MMAX/128));
    k_gemm<<<g, 256, 0, stream>>>(A, Wm, bias, meta, K, N, NP, Cf, Cb, act);
  };

  k_setup<<<1, 32, 0, stream>>>(tlens, meta);
  k_pack<<<dim3(NMAX, Lsz), 256, 0, stream>>>(emb, amask, meta, xf, xb, maskp);
  k_embmean<<<NMAX, 256, 0, stream>>>(xf, meta, emean);

  for (int s = 0; s < 2; ++s) {
    for (int i = 0; i < NLn; ++i) {
      gemm(xb, wQ[s] + (size_t)i*3*Dsz*Dsz, bQ[s] + i*3*Dsz, Dsz, 3*Dsz, 3*Dsz, nullptr, qkvb, 0);
      k_attn<<<dim3(Lsz*Hsz, NMAX/16), 32, 0, stream>>>(qkvb, maskp, meta, ob);
      gemm(ob, wO[s] + (size_t)i*Dsz*Dsz, bO[s] + i*Dsz, Dsz, Dsz, Dsz, af, nullptr, 0);
      k_lnres<<<MMAX, 256, 0, stream>>>(xf, af, g1[s] + i*Dsz, b1[s] + i*Dsz, meta, xf, xb);
      gemm(xb, w1b[s] + (size_t)i*Fsz*Dsz, c1[s] + i*Fsz, Dsz, Fsz, Fsz, nullptr, fb, 1);
      gemm(fb, w2b[s] + (size_t)i*Dsz*Fsz, c2[s] + i*Dsz, Fsz, Dsz, Dsz, af, nullptr, 0);
      k_lnres<<<MMAX, 256, 0, stream>>>(xf, af, g2[s] + i*Dsz, b2[s] + i*Dsz, meta, xf, xb);
    }
    if (s == 0) {   // bottleneck: z = tanh(h pe^T + b); zd = tanh(z pd^T + b)
      gemm(xb, pewb, pe_b, Dsz,  Zsz, ZPsz, nullptr, zb, 2);
      gemm(zb, pdwb, pd_b, ZPsz, Dsz, Dsz,  xf, xb, 2);
    }
  }

  k_score<<<NMAX, 256, 0, stream>>>(xf, emean, meta, scores);
  k_select<<<Bsz, 256, 0, stream>>>(scores, meta);

  float* o_oh   = (float*)d_out;
  float* o_ext  = o_oh  + (size_t)Bsz*Tsz;
  float* o_nadv = o_ext + (size_t)Bsz*Tsz*Lsz;
  float* o_ntl  = o_nadv + 1;
  float* o_emb  = o_ntl + Bsz;
  float* o_msk  = o_emb + (size_t)Bsz*Tsz*Lsz*Dsz;

  k_outsmall<<<Bsz, 256, 0, stream>>>(meta, o_oh, o_ext, o_nadv, o_ntl);
  k_gather<<<dim3(Bsz*Tsz, Lsz), 256, 0, stream>>>(meta, emb, amask, o_emb, o_msk);
}
